// GATLayer_31052613550680
// MI455X (gfx1250) — compile-verified, matrix operand order checked by reading, not run on hardware
//
#include <hip/hip_runtime.h>
#include <hip/hip_bf16.h>
#include <math.h>

typedef __attribute__((ext_vector_type(16))) __bf16 v16bf;
typedef __attribute__((ext_vector_type(8)))  float  v8f;

#define N_NODES_C 100000
#define HEADS_C   8
#define NEG_SLOPE_C 0.2f

// ---------- helpers ----------

__device__ __forceinline__ __bf16 f2bf(float f) {
  union { float f; unsigned u; } x; x.f = f;
  // round-to-nearest-even truncation to bf16
  unsigned r = (x.u + 0x7FFFu + ((x.u >> 16) & 1u)) >> 16;
  unsigned short s = (unsigned short)r;
  return *reinterpret_cast<__bf16*>(&s);
}

__device__ __forceinline__ void atomicMaxF(float* addr, float val) {
  // exact float max via integer atomics (works with -inf init)
  if (val >= 0.0f) atomicMax((int*)addr, __float_as_int(val));
  else             atomicMin((unsigned int*)addr, (unsigned int)__float_as_int(val));
}

// ---------- 0: init output accumulator / softmax state ----------

__global__ void gat_init(float* __restrict__ out, float* __restrict__ smax,
                         float* __restrict__ ssum, int n_out, int n_nh) {
  int i = blockIdx.x * blockDim.x + threadIdx.x;
  if (i < n_out) out[i] = 0.0f;
  if (i < n_nh) { smax[i] = -INFINITY; ssum[i] = 0.0f; }
}

// ---------- 1: h = feats @ W via v_wmma_f32_16x16x32_bf16 ----------
// Block: 256 threads (8 wave32), covers 64 rows. LDS: WT (128x128 bf16, transposed)
// + A tile (64x128 bf16) = 48KB.

__global__ void __launch_bounds__(256)
gat_proj_wmma(const float* __restrict__ feats, const float* __restrict__ W,
              float* __restrict__ h, int n_nodes) {
  __shared__ __bf16 WT[128 * 128];  // WT[n*128 + k]  (B operand, transposed)
  __shared__ __bf16 AT[64 * 128];   // AT[row*128 + k]

  const int tid = threadIdx.x;
  const int rowBase = blockIdx.x * 64;

  // Stage W transposed as bf16 (coalesced global reads).
  #pragma unroll
  for (int i = 0; i < 64; ++i) {
    int idx = tid + i * 256;           // idx = k*128 + n
    int k = idx >> 7, n = idx & 127;
    WT[n * 128 + k] = f2bf(W[idx]);
  }
  // Stage A tile (64 rows) as bf16, zero-pad out-of-range rows.
  #pragma unroll
  for (int i = 0; i < 32; ++i) {
    int idx = tid + i * 256;           // idx = r*128 + c
    int r = idx >> 7, c = idx & 127;
    int g = rowBase + r;
    AT[idx] = (g < n_nodes) ? f2bf(feats[(size_t)g * 128 + c]) : f2bf(0.0f);
  }
  __syncthreads();

  const int wv   = tid >> 5;     // wave id 0..7 (wave32)
  const int lane = tid & 31;
  const int mn   = lane & 15;    // M (for A) / N (for B) index
  const int half = lane >> 4;    // 0: lanes 0-15, 1: lanes 16-31
  const int rg     = wv >> 1;    // row group 0..3  (16 rows each)
  const int nhalf  = wv & 1;     // column half: 4 N-tiles each

  v8f c[4];
  #pragma unroll
  for (int t = 0; t < 4; ++t) c[t] = (v8f){0, 0, 0, 0, 0, 0, 0, 0};

  const int arow = (rg * 16 + mn) * 128;

  #pragma unroll
  for (int ks = 0; ks < 4; ++ks) {
    const int kb = ks * 32;
    // A fragment: ISA 16-bit 16x32 layout.
    v16bf a;
    #pragma unroll
    for (int i = 0; i < 8; ++i) {
      a[i]     = AT[arow + kb + half * 8 + i];
      a[8 + i] = AT[arow + kb + 16 + half * 8 + i];
    }
    #pragma unroll
    for (int t = 0; t < 4; ++t) {
      const int nb = (nhalf * 4 + t) * 16;
      const int bcol = (nb + mn) * 128;
      v16bf b;
      #pragma unroll
      for (int i = 0; i < 8; ++i) {
        b[i]     = WT[bcol + kb + half * 8 + i];
        b[8 + i] = WT[bcol + kb + 16 + half * 8 + i];
      }
      c[t] = __builtin_amdgcn_wmma_f32_16x16x32_bf16(
          false, a, false, b, (short)0, c[t], false, false);
    }
  }

  // D layout: VGPR r holds row (half*8 + r), column mn. Contiguous 64B stores.
  #pragma unroll
  for (int t = 0; t < 4; ++t) {
    const int nb = (nhalf * 4 + t) * 16;
    #pragma unroll
    for (int r = 0; r < 8; ++r) {
      int grow = rowBase + rg * 16 + half * 8 + r;
      if (grow < n_nodes) h[(size_t)grow * 128 + nb + mn] = c[t][r];
    }
  }
}

// ---------- 2: per-node attention halves el/er ----------

__global__ void gat_attn_coeff(const float* __restrict__ h,
                               const float* __restrict__ attn_l,
                               const float* __restrict__ attn_r,
                               float* __restrict__ el, float* __restrict__ er,
                               int n_nodes) {
  int idx = blockIdx.x * blockDim.x + threadIdx.x;  // node*8 + head
  if (idx >= n_nodes * HEADS_C) return;
  int n = idx >> 3, hd = idx & 7;
  const float* hp = h + (size_t)n * 128 + hd * 16;
  const float* al = attn_l + hd * 16;
  const float* ar = attn_r + hd * 16;
  float sl = 0.0f, sr = 0.0f;
  #pragma unroll
  for (int d = 0; d < 16; ++d) {
    float v = hp[d];
    sl = fmaf(v, al[d], sl);
    sr = fmaf(v, ar[d], sr);
  }
  el[idx] = sl;
  er[idx] = sr;
}

// ---------- 3: segment max over edges ----------

__global__ void gat_edge_max(const float* __restrict__ el, const float* __restrict__ er,
                             const int* __restrict__ src, const int* __restrict__ dst,
                             float* __restrict__ smax, int n_edges) {
  int e = blockIdx.x * blockDim.x + threadIdx.x;
  if (e >= n_edges) return;
  int s0 = src[e] * HEADS_C, d0 = dst[e] * HEADS_C;
  #pragma unroll
  for (int hd = 0; hd < HEADS_C; ++hd) {
    float sc = el[s0 + hd] + er[d0 + hd];
    sc = (sc > 0.0f) ? sc : NEG_SLOPE_C * sc;  // LeakyReLU
    atomicMaxF(&smax[d0 + hd], sc);
  }
}

// ---------- 4: segment sum of exp ----------

__global__ void gat_edge_sum(const float* __restrict__ el, const float* __restrict__ er,
                             const int* __restrict__ src, const int* __restrict__ dst,
                             const float* __restrict__ smax, float* __restrict__ ssum,
                             int n_edges) {
  int e = blockIdx.x * blockDim.x + threadIdx.x;
  if (e >= n_edges) return;
  int s0 = src[e] * HEADS_C, d0 = dst[e] * HEADS_C;
  #pragma unroll
  for (int hd = 0; hd < HEADS_C; ++hd) {
    float sc = el[s0 + hd] + er[d0 + hd];
    sc = (sc > 0.0f) ? sc : NEG_SLOPE_C * sc;
    atomicAdd(&ssum[d0 + hd], __expf(sc - smax[d0 + hd]));
  }
}

// ---------- 5: weighted gather-scatter aggregation ----------
// One thread per (edge, head): 64B gather of h[src], 16 f32 atomic adds into out[dst].

__global__ void gat_edge_aggr(const float* __restrict__ el, const float* __restrict__ er,
                              const float* __restrict__ smax, const float* __restrict__ ssum,
                              const float* __restrict__ h,
                              const int* __restrict__ src, const int* __restrict__ dst,
                              float* __restrict__ out, int n_edges) {
  int idx = blockIdx.x * blockDim.x + threadIdx.x;  // edge*8 + head
  if (idx >= n_edges * HEADS_C) return;
  int e = idx >> 3, hd = idx & 7;
  int s = src[e], d = dst[e];
  float sc = el[s * HEADS_C + hd] + er[d * HEADS_C + hd];
  sc = (sc > 0.0f) ? sc : NEG_SLOPE_C * sc;
  float alpha = __expf(sc - smax[d * HEADS_C + hd]) / ssum[d * HEADS_C + hd];
  const float4* hs = (const float4*)(h + (size_t)s * 128 + hd * 16);
  float* o = out + (size_t)d * 128 + hd * 16;
  #pragma unroll
  for (int q = 0; q < 4; ++q) {
    float4 v = hs[q];
    atomicAdd(o + q * 4 + 0, alpha * v.x);
    atomicAdd(o + q * 4 + 1, alpha * v.y);
    atomicAdd(o + q * 4 + 2, alpha * v.z);
    atomicAdd(o + q * 4 + 3, alpha * v.w);
  }
}

// ---------- 6: residual + bias ----------

__global__ void gat_final(float* __restrict__ out, const float* __restrict__ h,
                          const float* __restrict__ bias, int n_out) {
  int i = blockIdx.x * blockDim.x + threadIdx.x;
  if (i >= n_out) return;
  out[i] += h[i] + bias[i & 127];
}

// ---------- launch ----------

extern "C" void kernel_launch(void* const* d_in, const int* in_sizes, int n_in,
                              void* d_out, int out_size, void* d_ws, size_t ws_size,
                              hipStream_t stream) {
  const float* feats  = (const float*)d_in[0];
  const float* W      = (const float*)d_in[1];
  const float* attn_l = (const float*)d_in[2];
  const float* attn_r = (const float*)d_in[3];
  const float* bias   = (const float*)d_in[4];
  const int*   src    = (const int*)d_in[5];
  const int*   dst    = (const int*)d_in[6];
  float* out = (float*)d_out;

  const int n_nodes = (int)(in_sizes[0] / 128);  // 100000
  const int n_edges = in_sizes[5];               // 1600000
  const int n_out   = n_nodes * 128;
  const int n_nh    = n_nodes * HEADS_C;

  // Workspace layout (floats): h | el | er | smax | ssum  (~64 MB total)
  float* h    = (float*)d_ws;
  float* el   = h    + (size_t)n_nodes * 128;
  float* er   = el   + (size_t)n_nh;
  float* smax = er   + (size_t)n_nh;
  float* ssum = smax + (size_t)n_nh;

  gat_init<<<(n_out + 255) / 256, 256, 0, stream>>>(out, smax, ssum, n_out, n_nh);
  gat_proj_wmma<<<(n_nodes + 63) / 64, 256, 0, stream>>>(feats, W, h, n_nodes);
  gat_attn_coeff<<<(n_nh + 255) / 256, 256, 0, stream>>>(h, attn_l, attn_r, el, er, n_nodes);
  gat_edge_max<<<(n_edges + 255) / 256, 256, 0, stream>>>(el, er, src, dst, smax, n_edges);
  gat_edge_sum<<<(n_edges + 255) / 256, 256, 0, stream>>>(el, er, src, dst, smax, ssum, n_edges);
  gat_edge_aggr<<<(n_edges * HEADS_C + 255) / 256, 256, 0, stream>>>(
      el, er, smax, ssum, h, src, dst, out, n_edges);
  gat_final<<<(n_out + 255) / 256, 256, 0, stream>>>(out, h, bias, n_out);
}
